// MoEMLP_16930761081089
// MI455X (gfx1250) — matrix operand order repaired and checked by reading, not compiled, not run on hardware
//
#include <hip/hip_runtime.h>

#define C_DIM 1024
#define H_DIM 2048
#define E_DIM 8
#define N_TOK 2048
#define XP (C_DIM + 8)   // padded LDS row stride (bf16 elems), keeps 16B align + bank spread
#define HP (H_DIM + 8)

typedef __bf16 bf16;
typedef __attribute__((ext_vector_type(8)))  bf16  v8bf;
typedef __attribute__((ext_vector_type(16))) bf16  v16bf;
typedef __attribute__((ext_vector_type(8)))  float v8f;

#define WMMA_BF16(A, B, Cacc) \
    __builtin_amdgcn_wmma_f32_16x16x32_bf16(false, (A), false, (B), (short)0, (Cacc), false, false)

// ---- WMMA fragment loaders (wave32, 16-bit operands) ----------------------
// A 16x32 bf16 layout (ISA 7.12.2): lane<16 -> M=lane, elems K = {kb+0..7, kb+16..23}
//                                   lane>=16 -> M=lane-16, elems K = {kb+8..15, kb+24..31}
__device__ __forceinline__ v16bf load_fragA(const bf16* __restrict__ rowp, int kb, int hi) {
    const bf16* p = rowp + kb + hi * 8;
    v8bf lo = *(const v8bf*)p;
    v8bf hh = *(const v8bf*)(p + 16);
    return __builtin_shufflevector(lo, hh, 0,1,2,3,4,5,6,7,8,9,10,11,12,13,14,15);
}
// B 32x16 bf16 layout (matches A): lane<16 -> N=lane, K = {kb+0..7, kb+16..23}
//                                  lane>=16 -> N=lane-16, K = {kb+8..15, kb+24..31}
__device__ __forceinline__ v16bf load_fragB(const bf16* __restrict__ rowp, int kb, int hi) {
    const bf16* p = rowp + kb + hi * 8;
    v8bf lo = *(const v8bf*)p;
    v8bf hh = *(const v8bf*)(p + 16);
    return __builtin_shufflevector(lo, hh, 0,1,2,3,4,5,6,7,8,9,10,11,12,13,14,15);
}

// ---- weight prep: transpose [R][Cc] f32 -> [Cc][R] bf16, batched over E ----
__global__ __launch_bounds__(256)
void transpose_to_bf16(const float* __restrict__ in, bf16* __restrict__ out, int R, int Cc) {
    __shared__ float tile[32][33];
    const int e = blockIdx.z;
    const float* inE = in + (size_t)e * R * Cc;
    bf16* outE = out + (size_t)e * R * Cc;
    const int c0 = blockIdx.x * 32, r0 = blockIdx.y * 32;
    const int tx = threadIdx.x, ty = threadIdx.y;          // 32 x 8
    #pragma unroll
    for (int i = 0; i < 32; i += 8)
        tile[ty + i][tx] = inE[(size_t)(r0 + ty + i) * Cc + c0 + tx];
    __syncthreads();
    #pragma unroll
    for (int i = 0; i < 32; i += 8)
        outE[(size_t)(c0 + ty + i) * R + r0 + tx] = (bf16)tile[tx][ty + i];
}

// ---- gating: one wave per token, fp32 softmax, top-2 of experts 1..7 ------
__global__ __launch_bounds__(256)
void gate_kernel(const float* __restrict__ x, const float* __restrict__ gw,
                 const float* __restrict__ gb, float* __restrict__ cw) {
    const int token = blockIdx.x * 8 + (threadIdx.x >> 5);
    const int lane = threadIdx.x & 31;
    float acc[E_DIM];
    #pragma unroll
    for (int e = 0; e < E_DIM; ++e) acc[e] = 0.f;
    const float* xr = x + (size_t)token * C_DIM;
    for (int c = lane; c < C_DIM; c += 32) {
        float xv = xr[c];
        const float* gr = gw + (size_t)c * E_DIM;
        #pragma unroll
        for (int e = 0; e < E_DIM; ++e) acc[e] += xv * gr[e];
    }
    #pragma unroll
    for (int off = 16; off > 0; off >>= 1) {
        #pragma unroll
        for (int e = 0; e < E_DIM; ++e) acc[e] += __shfl_down(acc[e], off, 32);
    }
    if (lane == 0) {
        float l[E_DIM], mx = -1e30f;
        #pragma unroll
        for (int e = 0; e < E_DIM; ++e) { l[e] = acc[e] + gb[e]; mx = fmaxf(mx, l[e]); }
        float s = 0.f;
        #pragma unroll
        for (int e = 0; e < E_DIM; ++e) { l[e] = __expf(l[e] - mx); s += l[e]; }
        float inv = 1.f / s;
        #pragma unroll
        for (int e = 0; e < E_DIM; ++e) l[e] *= inv;
        int i1 = 1; float v1 = l[1];
        #pragma unroll
        for (int e = 2; e < E_DIM; ++e) if (l[e] > v1) { v1 = l[e]; i1 = e; }
        int i2 = 1; float v2 = -1.f;
        #pragma unroll
        for (int e = 1; e < E_DIM; ++e) if (e != i1 && l[e] > v2) { v2 = l[e]; i2 = e; }
        float cwr[E_DIM];
        #pragma unroll
        for (int e = 0; e < E_DIM; ++e) cwr[e] = 0.f;
        cwr[0] = l[0]; cwr[i1] = v1; cwr[i2] = v2;   // ROUTE_SCALE = 1
        float* dst = cw + (size_t)token * E_DIM;
        #pragma unroll
        for (int e = 0; e < E_DIM; ++e) dst[e] = cwr[e];
    }
}

// ---- fused per-expert SwiGLU + combine ------------------------------------
// 32 tokens/block, 512 threads (16 waves, 4/SIMD => ~256 VGPR budget, no spills).
// gridDim.y=2 splits experts (4 each); halves write partials summed later.
// Every B fragment feeds two WMMAs (two M-tiles) -> halved L2 weight traffic.
// N-columns processed in pairs of 16-wide tiles with a full K-pass per pair to
// keep the live accumulator set at 8 v8f (GEMM1) / 4 v8f (GEMM2).
__global__ __launch_bounds__(512)
void moe_expert_kernel(const float* __restrict__ x,
                       const bf16* __restrict__ wt1,   // [E][H][C] bf16
                       const bf16* __restrict__ wt3,   // [E][H][C] bf16
                       const bf16* __restrict__ wt2,   // [E][C][H] bf16
                       const float* __restrict__ b1,   // [E][H]
                       const float* __restrict__ b2,   // [E][C]
                       const float* __restrict__ b3,   // [E][H]
                       const float* __restrict__ cw,   // [N][E]
                       float* __restrict__ pbase) {    // 2 partial outputs [N][C]
    __shared__ bf16 xb[32][XP];
    __shared__ bf16 hb[32][HP];
    __shared__ float cwt[32][E_DIM];

    const int tid  = threadIdx.x;
    const int wave = tid >> 5;          // 0..15
    const int lane = tid & 31;
    const int row  = lane & 15;
    const int hi   = lane >> 4;
    const int m0   = blockIdx.x * 32;
    const int e0   = blockIdx.y * 4;
    float* pout = pbase + (size_t)blockIdx.y * N_TOK * C_DIM;

    // stage x tile as bf16
    for (int i = tid; i < 32 * C_DIM; i += 512) {
        int m = i >> 10, c = i & (C_DIM - 1);
        xb[m][c] = (bf16)x[(size_t)(m0 + m) * C_DIM + c];
    }
    if (tid < 32 * E_DIM) {
        int m = tid >> 3, e = tid & 7;
        cwt[m][e] = cw[(size_t)(m0 + m) * E_DIM + e];
    }

    v8f oacc[2][4] = {};   // [m-tile][n-tile]; wave owns 64 out cols, persists over experts

    for (int ee = 0; ee < 4; ++ee) {
        const int e = e0 + ee;
        const bf16* w1e = wt1 + (size_t)e * H_DIM * C_DIM;
        const bf16* w3e = wt3 + (size_t)e * H_DIM * C_DIM;
        __syncthreads();   // prior GEMM2 readers of hb done (also covers xb staging)

        // ---- GEMM1: a = x@w1, g = x@w3; wave owns 128 h-cols, 4 pair-passes
        for (int p = 0; p < 4; ++p) {
            const int nb = wave * 128 + p * 32;   // 2 tiles: nb, nb+16
            v8f a1[2][2] = {}, a3[2][2] = {};
            for (int ks = 0; ks < C_DIM / 32; ++ks) {
                const int kb = ks * 32;
                v16bf af0 = load_fragA(&xb[row][0], kb, hi);
                v16bf af1 = load_fragA(&xb[16 + row][0], kb, hi);
                #pragma unroll
                for (int t = 0; t < 2; ++t) {
                    const size_t roff = (size_t)(nb + t * 16 + row) * C_DIM;
                    v16bf bf1 = load_fragB(w1e + roff, kb, hi);
                    a1[0][t] = WMMA_BF16(af0, bf1, a1[0][t]);
                    a1[1][t] = WMMA_BF16(af1, bf1, a1[1][t]);
                    v16bf bf3 = load_fragB(w3e + roff, kb, hi);
                    a3[0][t] = WMMA_BF16(af0, bf3, a3[0][t]);
                    a3[1][t] = WMMA_BF16(af1, bf3, a3[1][t]);
                }
            }
            // ---- SwiGLU epilogue for this pair -> hb (bf16) ----
            #pragma unroll
            for (int mt = 0; mt < 2; ++mt) {
                #pragma unroll
                for (int t = 0; t < 2; ++t) {
                    const int n = nb + t * 16 + row;
                    const float bb1 = b1[(size_t)e * H_DIM + n];
                    const float bb3 = b3[(size_t)e * H_DIM + n];
                    #pragma unroll
                    for (int j = 0; j < 8; ++j) {
                        const int m = 16 * mt + j + 8 * hi;
                        float av = a1[mt][t][j] + bb1;
                        float gv = a3[mt][t][j] + bb3;
                        hb[m][n] = (bf16)((gv / (1.f + __expf(-gv))) * av);
                    }
                }
            }
        }
        __syncthreads();
        // ---- GEMM2: h@w2; wave owns 64 out-cols, 2 pair-passes ----
        const bf16* w2e = wt2 + (size_t)e * C_DIM * H_DIM;
        for (int q = 0; q < 2; ++q) {
            const int cb = wave * 64 + q * 32;
            v8f ps[2][2] = {};
            for (int ks = 0; ks < H_DIM / 32; ++ks) {
                const int kb = ks * 32;
                v16bf af0 = load_fragA(&hb[row][0], kb, hi);
                v16bf af1 = load_fragA(&hb[16 + row][0], kb, hi);
                #pragma unroll
                for (int t = 0; t < 2; ++t) {
                    v16bf bfb = load_fragB(w2e + (size_t)(cb + t * 16 + row) * H_DIM, kb, hi);
                    ps[0][t] = WMMA_BF16(af0, bfb, ps[0][t]);
                    ps[1][t] = WMMA_BF16(af1, bfb, ps[1][t]);
                }
            }
            #pragma unroll
            for (int mt = 0; mt < 2; ++mt) {
                #pragma unroll
                for (int t = 0; t < 2; ++t) {
                    const int n = cb + t * 16 + row;
                    const float bb2 = b2[(size_t)e * C_DIM + n];
                    #pragma unroll
                    for (int j = 0; j < 8; ++j) {
                        float w = cwt[16 * mt + j + 8 * hi][e];
                        oacc[mt][q * 2 + t][j] += w * (ps[mt][t][j] + bb2);
                    }
                }
            }
        }
    }
    // ---- writeback partial ----
    #pragma unroll
    for (int mt = 0; mt < 2; ++mt) {
        #pragma unroll
        for (int tt = 0; tt < 4; ++tt) {
            const int n = wave * 64 + tt * 16 + row;
            #pragma unroll
            for (int j = 0; j < 8; ++j)
                pout[(size_t)(m0 + 16 * mt + j + 8 * hi) * C_DIM + n] = oacc[mt][tt][j];
        }
    }
}

// ---- deterministic sum of the two expert-half partials --------------------
__global__ __launch_bounds__(256)
void combine_kernel(const float* __restrict__ p0, const float* __restrict__ p1,
                    float* __restrict__ out) {
    const size_t i = (size_t)blockIdx.x * 256 + threadIdx.x;   // float4 index
    const float4 a = ((const float4*)p0)[i];
    const float4 b = ((const float4*)p1)[i];
    float4 o; o.x = a.x + b.x; o.y = a.y + b.y; o.z = a.z + b.z; o.w = a.w + b.w;
    ((float4*)out)[i] = o;
}

// ---- deterministic balance loss (selected <=> cw > 0, softmax is positive) -
__global__ __launch_bounds__(256)
void loss_kernel(const float* __restrict__ cw, float* __restrict__ out_loss) {
    __shared__ float su[256 * E_DIM];
    __shared__ float ss[256 * E_DIM];
    const int tid = threadIdx.x;
    float u[E_DIM], s[E_DIM];
    #pragma unroll
    for (int e = 0; e < E_DIM; ++e) { u[e] = 0.f; s[e] = 0.f; }
    for (int t = tid; t < N_TOK; t += 256) {
        #pragma unroll
        for (int e = 0; e < E_DIM; ++e) {
            float c = cw[(size_t)t * E_DIM + e];
            if (c > 0.f) { u[e] += 1.f; s[e] += c; }
        }
    }
    #pragma unroll
    for (int e = 0; e < E_DIM; ++e) { su[tid * E_DIM + e] = u[e]; ss[tid * E_DIM + e] = s[e]; }
    __syncthreads();
    for (int off = 128; off > 0; off >>= 1) {
        if (tid < off) {
            #pragma unroll
            for (int e = 0; e < E_DIM; ++e) {
                su[tid * E_DIM + e] += su[(tid + off) * E_DIM + e];
                ss[tid * E_DIM + e] += ss[(tid + off) * E_DIM + e];
            }
        }
        __syncthreads();
    }
    if (tid == 0) {
        const float coef = (float)E_DIM / (3.f * (float)N_TOK);   // E/(Kp*N), Kp=3
        float loss = 0.f;
        #pragma unroll
        for (int e = 0; e < E_DIM; ++e) loss += coef * su[e] * (ss[e] / (float)N_TOK);
        out_loss[0] = loss;
    }
}

extern "C" void kernel_launch(void* const* d_in, const int* in_sizes, int n_in,
                              void* d_out, int out_size, void* d_ws, size_t ws_size,
                              hipStream_t stream) {
    (void)in_sizes; (void)n_in; (void)out_size; (void)ws_size;
    const float* x  = (const float*)d_in[0];
    const float* gw = (const float*)d_in[1];
    const float* gb = (const float*)d_in[2];
    const float* w1 = (const float*)d_in[3];
    const float* b1 = (const float*)d_in[4];
    const float* w2 = (const float*)d_in[5];
    const float* b2 = (const float*)d_in[6];
    const float* w3 = (const float*)d_in[7];
    const float* b3 = (const float*)d_in[8];
    float* out = (float*)d_out;

    char* ws = (char*)d_ws;
    const size_t wbytes = (size_t)E_DIM * C_DIM * H_DIM * sizeof(bf16);  // 32 MB each
    const size_t pbytes = (size_t)N_TOK * C_DIM * sizeof(float);         // 8 MB each
    bf16* wt1   = (bf16*)(ws);
    bf16* wt3   = (bf16*)(ws + wbytes);
    bf16* wt2   = (bf16*)(ws + 2 * wbytes);
    float* pout = (float*)(ws + 3 * wbytes);            // two partials back-to-back
    float* cw   = (float*)(ws + 3 * wbytes + 2 * pbytes);

    dim3 tb(32, 8);
    transpose_to_bf16<<<dim3(H_DIM / 32, C_DIM / 32, E_DIM), tb, 0, stream>>>(w1, wt1, C_DIM, H_DIM);
    transpose_to_bf16<<<dim3(H_DIM / 32, C_DIM / 32, E_DIM), tb, 0, stream>>>(w3, wt3, C_DIM, H_DIM);
    transpose_to_bf16<<<dim3(C_DIM / 32, H_DIM / 32, E_DIM), tb, 0, stream>>>(w2, wt2, H_DIM, C_DIM);

    gate_kernel<<<N_TOK / 8, 256, 0, stream>>>(x, gw, gb, cw);
    moe_expert_kernel<<<dim3(N_TOK / 32, 2), 512, 0, stream>>>(
        x, wt1, wt3, wt2, b1, b2, b3, cw, pout);
    combine_kernel<<<(N_TOK * C_DIM / 4) / 256, 256, 0, stream>>>(
        pout, pout + N_TOK * C_DIM, out);
    loss_kernel<<<1, 256, 0, stream>>>(cw, out + (size_t)N_TOK * C_DIM);
}
